// CotNetLayer_56908316672431
// MI455X (gfx1250) — matrix operand order repaired and checked by reading, not compiled, not run on hardware
//
#include <hip/hip_runtime.h>
#include <math.h>

// ---------- types for WMMA ----------
typedef __attribute__((ext_vector_type(16))) __bf16 v16bf;
typedef __attribute__((ext_vector_type(8)))  float  v8f;
typedef __attribute__((ext_vector_type(4)))  unsigned u32x4;

#define BB 4
#define HH 128
#define WW 128
#define CC 256
#define M_TOT (BB*HH*WW)     // 65536 pixels
#define K2 9
#define GG 32                // C / SP
#define SP 8
#define NW2 288              // K2 * C / SP
#define CH 128               // C / 2
#define EPS_BN 1e-3f
#define EPS_GN 1e-5f

// ---------- bf16 helpers (bit-level, RNE) ----------
__device__ __forceinline__ unsigned short f2bf(float f) {
  unsigned u = __float_as_uint(f);
  u += 0x7fffu + ((u >> 16) & 1u);
  return (unsigned short)(u >> 16);
}
__device__ __forceinline__ float bf2f(unsigned short h) {
  return __uint_as_float(((unsigned)h) << 16);
}

// ---------- WMMA fragments as two 128-bit loads ----------
union Frag { v16bf v; u32x4 q[2]; };

// A (16x32 bf16): lane half selects K blocks; dwords 0-3 = K 8h..8h+7,
// dwords 4-7 = K 16+8h..16+8h+7 -> two contiguous 16B runs.
__device__ __forceinline__ void load_a(Frag& f, const unsigned short* p, int half) {
  f.q[0] = *(const u32x4*)(p + (half << 3));
  f.q[1] = *(const u32x4*)(p + 16 + (half << 3));
}
// B (32x16 bf16) with weights stored [N][K]: dwords 0-3 = K 8h..8h+7,
// dwords 4-7 = K 16+8h..16+8h+7 (same order as A).
__device__ __forceinline__ void load_b(Frag& f, const unsigned short* p, int half) {
  f.q[0] = *(const u32x4*)(p + (half << 3));
  f.q[1] = *(const u32x4*)(p + 16 + (half << 3));
}
__device__ __forceinline__ void zero_frag(Frag& f) {
  f.q[0] = (u32x4){0u, 0u, 0u, 0u};
  f.q[1] = (u32x4){0u, 0u, 0u, 0u};
}
#define WMMA_BF16(A, B, C) \
  __builtin_amdgcn_wmma_f32_16x16x32_bf16(false, (A), false, (B), (short)0, (C), false, false)

__device__ __forceinline__ float block_reduce(float v, float* sm) {
  int tid = threadIdx.x;
  sm[tid] = v; __syncthreads();
  for (int s = blockDim.x >> 1; s > 0; s >>= 1) {
    if (tid < s) sm[tid] += sm[tid + s];
    __syncthreads();
  }
  float r = sm[0]; __syncthreads();
  return r;
}

// ---------- conversion kernels ----------
__global__ void cvt_f32_bf16(const float* __restrict__ in,
                             unsigned short* __restrict__ out, int n) {
  int i = blockIdx.x * blockDim.x + threadIdx.x;
  if (i < n) out[i] = f2bf(in[i]);
}
// weight [K][N] f32 -> [N][K] bf16 so B-fragment K runs are contiguous
__global__ void wt_transpose(const float* __restrict__ in,
                             unsigned short* __restrict__ out, int K, int N) {
  int i = blockIdx.x * blockDim.x + threadIdx.x;
  if (i < K * N) {
    int k = i / N, n = i % N;
    out[n * K + k] = f2bf(in[i]);
  }
}

// ---------- key: 3x3 conv C->C (+bias, BN, ReLU) -> bf16 ----------
// 64x16 per wave: 4 M-subtiles share each B (weight) fragment.
__global__ void keyconv_wmma(const unsigned short* __restrict__ xb,
                             const unsigned short* __restrict__ kwT, // [256][2304]
                             const float* __restrict__ key_b,
                             const float* __restrict__ bn_g,
                             const float* __restrict__ bn_b,
                             unsigned short* __restrict__ kb) {
  int wave = (blockIdx.x * blockDim.x + threadIdx.x) >> 5;
  int lane = threadIdx.x & 31;
  const int tiles_n = CC / 16;                       // 16
  int tm = (wave / tiles_n) << 6;                    // 64-row super-tile
  int tn = (wave % tiles_n) << 4;
  int half = lane >> 4, l16 = lane & 15;
  int n = tn + l16;
  int b = tm >> 14, y = (tm >> 7) & 127;
  int x0 = (tm & 127) + l16;                         // subtile i at x0 + 16*i
  const unsigned short* bbase = kwT + (size_t)n * (K2 * CC);
  __builtin_prefetch(bbase, 0, 3);
  v8f acc[4] = {};
  for (int tap = 0; tap < K2; ++tap) {
    int dy = tap / 3 - 1, dx = tap % 3 - 1;
    int y2 = y + dy;
    bool rowok = (unsigned)y2 < HH;
    const unsigned short* rowbase = xb + (size_t)((b * HH + y2) * WW) * CC;
    const unsigned short* btap = bbase + tap * CC;
    int x2 = x0 + dx;
    bool ok[4];
    const unsigned short* arow[4];
#pragma unroll
    for (int i = 0; i < 4; ++i) {
      int xi = x2 + (i << 4);
      ok[i] = rowok && ((unsigned)xi < WW);
      arow[i] = rowbase + (size_t)xi * CC;
    }
    for (int cb = 0; cb < CC; cb += 32) {
      Frag bf; load_b(bf, btap + cb, half);
#pragma unroll
      for (int i = 0; i < 4; ++i) {
        Frag af;
        if (ok[i]) load_a(af, arow[i] + cb, half); else zero_frag(af);
        acc[i] = WMMA_BF16(af.v, bf.v, acc[i]);
      }
    }
  }
  float scale = bn_g[n] * rsqrtf(1.0f + EPS_BN);
  float beta  = bn_b[n];
  float bias  = key_b[n];
#pragma unroll
  for (int i = 0; i < 4; ++i)
#pragma unroll
    for (int r = 0; r < 8; ++r) {
      int mr = tm + (i << 4) + r + (half << 3);
      float v = (acc[i][r] + bias) * scale + beta;
      kb[(size_t)mr * CC + n] = f2bf(v > 0.f ? v : 0.f);
    }
}

// ---------- w1: 1x1 conv [x|k] 512 -> 128, BN, ReLU -> bf16 ----------
__global__ void w1_wmma(const unsigned short* __restrict__ xb,
                        const unsigned short* __restrict__ kb,
                        const unsigned short* __restrict__ w1T,   // [128][512]
                        const float* __restrict__ bn_g,
                        const float* __restrict__ bn_b,
                        unsigned short* __restrict__ hb) {
  int wave = (blockIdx.x * blockDim.x + threadIdx.x) >> 5;
  int lane = threadIdx.x & 31;
  const int tiles_n = CH / 16;                       // 8
  int tm = (wave / tiles_n) << 6;
  int tn = (wave % tiles_n) << 4;
  int half = lane >> 4, l16 = lane & 15;
  int n = tn + l16;
  const unsigned short* brow = w1T + (size_t)n * (2 * CC);
  __builtin_prefetch(brow, 0, 3);
  const unsigned short* ax[4];
  const unsigned short* ak[4];
#pragma unroll
  for (int i = 0; i < 4; ++i) {
    size_t m = (size_t)(tm + (i << 4) + l16);
    ax[i] = xb + m * CC;
    ak[i] = kb + m * CC;
  }
  v8f acc[4] = {};
  for (int ks = 0; ks < 16; ++ks) {
    Frag bf; load_b(bf, brow + ks * 32, half);
#pragma unroll
    for (int i = 0; i < 4; ++i) {
      const unsigned short* asrc = (ks < 8) ? (ax[i] + ks * 32)
                                            : (ak[i] + (ks - 8) * 32);
      Frag af; load_a(af, asrc, half);
      acc[i] = WMMA_BF16(af.v, bf.v, acc[i]);
    }
  }
  float scale = bn_g[n] * rsqrtf(1.0f + EPS_BN);
  float beta  = bn_b[n];
#pragma unroll
  for (int i = 0; i < 4; ++i)
#pragma unroll
    for (int r = 0; r < 8; ++r) {
      int mr = tm + (i << 4) + r + (half << 3);
      float v = acc[i][r] * scale + beta;
      hb[(size_t)mr * CH + n] = f2bf(v > 0.f ? v : 0.f);
    }
}

// ---------- w2: 1x1 conv 128 -> 288 (+bias) -> f32 ----------
__global__ void w2_wmma(const unsigned short* __restrict__ hb,
                        const unsigned short* __restrict__ w2T,   // [288][128]
                        const float* __restrict__ w2_b,
                        float* __restrict__ wgn) {
  int wave = (blockIdx.x * blockDim.x + threadIdx.x) >> 5;
  int lane = threadIdx.x & 31;
  const int tiles_n = NW2 / 16;                      // 18
  int tm = (wave / tiles_n) << 6;
  int tn = (wave % tiles_n) << 4;
  int half = lane >> 4, l16 = lane & 15;
  int n = tn + l16;
  const unsigned short* brow = w2T + (size_t)n * CH;
  const unsigned short* arow[4];
#pragma unroll
  for (int i = 0; i < 4; ++i)
    arow[i] = hb + (size_t)(tm + (i << 4) + l16) * CH;
  v8f acc[4] = {};
  for (int ks = 0; ks < 4; ++ks) {
    Frag bf; load_b(bf, brow + ks * 32, half);
#pragma unroll
    for (int i = 0; i < 4; ++i) {
      Frag af; load_a(af, arow[i] + ks * 32, half);
      acc[i] = WMMA_BF16(af.v, bf.v, acc[i]);
    }
  }
  float bias = w2_b[n];
#pragma unroll
  for (int i = 0; i < 4; ++i)
#pragma unroll
    for (int r = 0; r < 8; ++r) {
      int mr = tm + (i << 4) + r + (half << 3);
      wgn[(size_t)mr * NW2 + n] = acc[i][r] + bias;
    }
}

// ---------- val: 1x1 conv C->C (+bias, BN, SiLU) -> bf16 ----------
__global__ void val_wmma(const unsigned short* __restrict__ xb,
                         const unsigned short* __restrict__ vwT,   // [256][256]
                         const float* __restrict__ val_b,
                         const float* __restrict__ bn_g,
                         const float* __restrict__ bn_b,
                         unsigned short* __restrict__ vb) {
  int wave = (blockIdx.x * blockDim.x + threadIdx.x) >> 5;
  int lane = threadIdx.x & 31;
  const int tiles_n = CC / 16;                       // 16
  int tm = (wave / tiles_n) << 6;
  int tn = (wave % tiles_n) << 4;
  int half = lane >> 4, l16 = lane & 15;
  int n = tn + l16;
  const unsigned short* brow = vwT + (size_t)n * CC;
  __builtin_prefetch(brow, 0, 3);
  const unsigned short* arow[4];
#pragma unroll
  for (int i = 0; i < 4; ++i)
    arow[i] = xb + (size_t)(tm + (i << 4) + l16) * CC;
  v8f acc[4] = {};
  for (int ks = 0; ks < 8; ++ks) {
    Frag bf; load_b(bf, brow + ks * 32, half);
#pragma unroll
    for (int i = 0; i < 4; ++i) {
      Frag af; load_a(af, arow[i] + ks * 32, half);
      acc[i] = WMMA_BF16(af.v, bf.v, acc[i]);
    }
  }
  float scale = bn_g[n] * rsqrtf(1.0f + EPS_BN);
  float beta  = bn_b[n];
  float bias  = val_b[n];
#pragma unroll
  for (int i = 0; i < 4; ++i)
#pragma unroll
    for (int r = 0; r < 8; ++r) {
      int mr = tm + (i << 4) + r + (half << 3);
      float t = (acc[i][r] + bias) * scale + beta;
      float s = t / (1.0f + expf(-t));               // SiLU
      vb[(size_t)mr * CC + n] = f2bf(s);
    }
}

// ---------- GroupNorm stats: per (b,g) mean / rsqrt(var+eps) ----------
__global__ void gn_stats(const float* __restrict__ wgn, float* __restrict__ stats) {
  __shared__ float sm1[256], sm2[256];
  int bg = blockIdx.x;                               // 0..127
  int b = bg >> 5, g = bg & 31;
  const float* base = wgn + (size_t)b * 16384 * NW2 + g * K2;
  float s = 0.f, s2 = 0.f;
  for (int p = threadIdx.x; p < 16384; p += blockDim.x) {
    const float* row = base + (size_t)p * NW2;
#pragma unroll
    for (int kk = 0; kk < K2; ++kk) { float v = row[kk]; s += v; s2 += v * v; }
  }
  float ts  = block_reduce(s,  sm1);
  float ts2 = block_reduce(s2, sm2);
  if (threadIdx.x == 0) {
    const float inv = 1.0f / (16384.0f * K2);
    float mu = ts * inv;
    float var = ts2 * inv - mu * mu;
    stats[bg * 2]     = mu;
    stats[bg * 2 + 1] = rsqrtf(var + EPS_GN);
  }
}

__global__ void gn_norm(float* __restrict__ wgn,
                        const float* __restrict__ stats,
                        const float* __restrict__ gn_g,
                        const float* __restrict__ gn_b) {
  size_t i = (size_t)blockIdx.x * blockDim.x + threadIdx.x;
  if (i >= (size_t)M_TOT * NW2) return;
  int m = (int)(i / NW2), c = (int)(i % NW2);
  int bg = ((m >> 14) << 5) + c / K2;
  float mu = stats[bg * 2], rs = stats[bg * 2 + 1];
  wgn[i] = (wgn[i] - mu) * rs * gn_g[c] + gn_b[c];
}

// ---------- unfold + dynamic local aggregation -> agg (f32, in d_out) ----------
__global__ void agg_kernel(const float* __restrict__ wgn,
                           const unsigned short* __restrict__ vb,
                           float* __restrict__ agg) {
  __shared__ float w9s[NW2];
  int m = blockIdx.x, c = threadIdx.x;
  w9s[c] = wgn[(size_t)m * NW2 + c];
  if (c < NW2 - 256) w9s[256 + c] = wgn[(size_t)m * NW2 + 256 + c];
  __syncthreads();
  int b = m >> 14, y = (m >> 7) & 127, x = m & 127;
  const float* wg = w9s + (c >> 3) * K2;
  float sum = 0.f;
#pragma unroll
  for (int kk = 0; kk < K2; ++kk) {
    int y2 = y + kk / 3 - 1, x2 = x + kk % 3 - 1;
    if ((unsigned)y2 < HH && (unsigned)x2 < WW)
      sum += bf2f(vb[(size_t)(((b * HH + y2) * WW) + x2) * CC + c]) * wg[kk];
  }
  agg[(size_t)m * CC + c] = sum;
}

// ---------- GAP over H,W of (agg + k) ----------
__global__ void gap_kernel(const float* __restrict__ agg,
                           const unsigned short* __restrict__ kb,
                           float* __restrict__ gap) {
  __shared__ float sm[256];
  int bc = blockIdx.x;                               // 0..1023
  int b = bc >> 8, c = bc & 255;
  float s = 0.f;
  for (int p = threadIdx.x; p < 16384; p += blockDim.x) {
    size_t idx = (size_t)(b * 16384 + p) * CC + c;
    s += agg[idx] + bf2f(kb[idx]);
  }
  float ts = block_reduce(s, sm);
  if (threadIdx.x == 0) gap[bc] = ts * (1.0f / 16384.0f);
}

// ---------- SE MLP + radix-2 softmax (tiny; single block) ----------
__global__ void se_kernel(const float* __restrict__ gap,
                          const float* __restrict__ se1_w, const float* __restrict__ se1_b,
                          const float* __restrict__ se1_bn_g, const float* __restrict__ se1_bn_b,
                          const float* __restrict__ se2_w, const float* __restrict__ se2_b,
                          float* __restrict__ attn) {
  __shared__ float gs[1024];
  __shared__ float a1[512];
  __shared__ float a2[2048];
  int tid = threadIdx.x;                             // block of 512
  for (int i = tid; i < 1024; i += 512) gs[i] = gap[i];
  __syncthreads();
  {
    int b = tid >> 7, co = tid & 127;
    float s = 0.f;
    for (int ci = 0; ci < 256; ++ci) s += gs[b * 256 + ci] * se1_w[ci * 128 + co];
    s = (s + se1_b[co]) * se1_bn_g[co] * rsqrtf(1.0f + EPS_BN) + se1_bn_b[co];
    a1[tid] = s > 0.f ? s : 0.f;
  }
  __syncthreads();
  for (int idx = tid; idx < 2048; idx += 512) {
    int b = idx >> 9, j = idx & 511;
    float s = 0.f;
    for (int ci = 0; ci < 128; ++ci) s += a1[b * 128 + ci] * se2_w[ci * 512 + j];
    a2[idx] = s + se2_b[j];
  }
  __syncthreads();
  for (int idx = tid; idx < 1024; idx += 512) {
    int b = idx >> 8, c = idx & 255;
    float x0 = a2[b * 512 + 2 * c], x1 = a2[b * 512 + 2 * c + 1];
    float mx = fmaxf(x0, x1);
    float e0 = expf(x0 - mx), e1 = expf(x1 - mx);
    float inv = 1.0f / (e0 + e1);
    attn[idx * 2]     = e0 * inv;
    attn[idx * 2 + 1] = e1 * inv;
  }
}

// ---------- final combine: out = agg*attn0 + k*attn1 (in place on d_out) ----------
__global__ void combine_kernel(float* __restrict__ out,
                               const unsigned short* __restrict__ kb,
                               const float* __restrict__ attn) {
  size_t i = (size_t)blockIdx.x * blockDim.x + threadIdx.x;
  if (i >= (size_t)M_TOT * CC) return;
  int c = (int)(i & 255);
  int b = (int)(i >> 22);                            // i / (16384*256)
  float a0 = attn[b * 512 + 2 * c], a1 = attn[b * 512 + 2 * c + 1];
  out[i] = out[i] * a0 + bf2f(kb[i]) * a1;
}

extern "C" void kernel_launch(void* const* d_in, const int* in_sizes, int n_in,
                              void* d_out, int out_size, void* d_ws, size_t ws_size,
                              hipStream_t stream) {
  const float* x        = (const float*)d_in[0];
  const float* key_w    = (const float*)d_in[1];
  const float* key_b    = (const float*)d_in[2];
  const float* key_bn_g = (const float*)d_in[3];
  const float* key_bn_b = (const float*)d_in[4];
  const float* w1_w     = (const float*)d_in[5];
  const float* w1_bn_g  = (const float*)d_in[6];
  const float* w1_bn_b  = (const float*)d_in[7];
  const float* w2_w     = (const float*)d_in[8];
  const float* w2_b     = (const float*)d_in[9];
  const float* gn_g     = (const float*)d_in[10];
  const float* gn_b     = (const float*)d_in[11];
  const float* val_w    = (const float*)d_in[12];
  const float* val_b    = (const float*)d_in[13];
  const float* val_bn_g = (const float*)d_in[14];
  const float* val_bn_b = (const float*)d_in[15];
  const float* se1_w    = (const float*)d_in[16];
  const float* se1_b    = (const float*)d_in[17];
  const float* se1_bn_g = (const float*)d_in[18];
  const float* se1_bn_b = (const float*)d_in[19];
  const float* se2_w    = (const float*)d_in[20];
  const float* se2_b    = (const float*)d_in[21];
  (void)in_sizes; (void)n_in; (void)out_size; (void)ws_size;

  char* ws = (char*)d_ws;
  unsigned short* xb    = (unsigned short*)(ws + 0);             // 32 MB
  unsigned short* kb    = (unsigned short*)(ws + 33554432);      // 32 MB
  unsigned short* hb    = (unsigned short*)(ws + 67108864);      // 16 MB
  float*          wgn   = (float*)         (ws + 83886080);      // 72 MB
  unsigned short* vb    = (unsigned short*)(ws + 159383552);     // 32 MB
  unsigned short* kwT   = (unsigned short*)(ws + 192937984);     // [256][2304]
  unsigned short* w1T   = (unsigned short*)(ws + 194117632);     // [128][512]
  unsigned short* w2T   = (unsigned short*)(ws + 194248704);     // [288][128]
  unsigned short* vwT   = (unsigned short*)(ws + 194322432);     // [256][256]
  float*          stats = (float*)         (ws + 194453504);     // [128][2]
  float*          gapb  = (float*)         (ws + 194454528);     // [4][256]
  float*          attnb = (float*)         (ws + 194458624);     // [4][256][2]
  float*          aggb  = (float*)d_out;                         // agg lives in d_out

  // 1. convert x and weights to bf16 (weights transposed to [N][K])
  cvt_f32_bf16<<<(M_TOT * CC) / 256, 256, 0, stream>>>(x, xb, M_TOT * CC);
  wt_transpose<<<(2304 * 256 + 255) / 256, 256, 0, stream>>>(key_w, kwT, 2304, 256);
  wt_transpose<<<(512 * 128 + 255) / 256, 256, 0, stream>>>(w1_w, w1T, 512, 128);
  wt_transpose<<<(128 * 288 + 255) / 256, 256, 0, stream>>>(w2_w, w2T, 128, 288);
  wt_transpose<<<(256 * 256 + 255) / 256, 256, 0, stream>>>(val_w, vwT, 256, 256);

  // 2. key 3x3 conv: (65536/64)*(256/16) = 16384 waves
  keyconv_wmma<<<2048, 256, 0, stream>>>(xb, kwT, key_b, key_bn_g, key_bn_b, kb);
  // 3. w1 1x1 over [x|k]: 8192 waves
  w1_wmma<<<1024, 256, 0, stream>>>(xb, kb, w1T, w1_bn_g, w1_bn_b, hb);
  // 4. w2 1x1 -> 288 dyn-weight logits: 18432 waves
  w2_wmma<<<2304, 256, 0, stream>>>(hb, w2T, w2_b, wgn);
  // 5. GroupNorm (32 groups of 9, stats over H,W,K2 per batch)
  gn_stats<<<128, 256, 0, stream>>>(wgn, stats);
  gn_norm<<<(M_TOT * NW2) / 256, 256, 0, stream>>>(wgn, stats, gn_g, gn_b);
  // 6. value 1x1 conv + SiLU: 16384 waves
  val_wmma<<<2048, 256, 0, stream>>>(xb, vwT, val_b, val_bn_g, val_bn_b, vb);
  // 7. unfold + aggregate -> d_out
  agg_kernel<<<M_TOT, 256, 0, stream>>>(wgn, vb, aggb);
  // 8. GAP of (agg + k)
  gap_kernel<<<1024, 256, 0, stream>>>(aggb, kb, gapb);
  // 9. SE MLP + radix softmax
  se_kernel<<<1, 512, 0, stream>>>(gapb, se1_b ? se1_w : se1_w, se1_b, se1_bn_g, se1_bn_b,
                                   se2_w, se2_b, attnb);
  // 10. combine in place
  combine_kernel<<<(M_TOT * CC) / 256, 256, 0, stream>>>(aggb, kb, attnb);
}
